// HPCrnn_67542655696955
// MI455X (gfx1250) — matrix operand order repaired and checked by reading, not compiled, not run on hardware
//
#include <hip/hip_runtime.h>

// ---------------------------------------------------------------------------
// HPC RNN (hippocampal model) for gfx1250 / MI455X.
//
// Single persistent kernel runs all 100 recurrent steps:
//   - 32 blocks x 256 threads (8 waves) == 256 waves == one wave per 16x64
//     output tile of the 256x1024x1024 GEMMs -> all blocks co-resident,
//     grid spin-barrier safe.
//   - Each block owns one 64-column N-panel; both weight panels (W1,W2) are
//     staged into LDS ONCE (264 KB of the WGP's 320 KB) and re-read by
//     ds_load_b128 for all 100 steps -> no repeated L2 weight traffic.
//   - v_wmma_f32_16x16x32_f16 (f16 in, f32 accumulate); BOTH operands
//     double-buffered: chunk k+1's A (global) and B (LDS) fragments are
//     issued before chunk k's WMMAs, so waits carry slack instead of
//     dscnt==0 stalls.
//   - elementwise sigmoid/relu/state updates fused into GEMM epilogues.
// ---------------------------------------------------------------------------

typedef _Float16 h16;
typedef __attribute__((ext_vector_type(16))) _Float16 v16h;
typedef __attribute__((ext_vector_type(8)))  float    v8f;

#define BS       256
#define ECNUM    1024
#define CA1NUM   1024
#define CA3NUM   1024
#define TLEN     100
#define NBLK     32          // persistent grid: 32 blocks x 256 threads
#define LDSTRIDE 1032        // padded LDS row stride (halves): 16B-aligned,
                             // lane i -> bank 4i mod 64 (conflict-free)
#define PANEL    (64 * LDSTRIDE)
#define LDS_BYTES (2 * PANEL * 2)

__device__ __forceinline__ float sigmoidf_(float x) {
    return 1.0f / (1.0f + __expf(-x));
}

// Grid-wide barrier: release fence -> block arrive -> generation spin ->
// acquire fence. All 32 blocks are co-resident (tiny grid), so this is safe.
__device__ __forceinline__ void grid_sync(unsigned* bar, unsigned* gen) {
    __builtin_amdgcn_fence(__ATOMIC_RELEASE, "agent");   // flush my stores
    __syncthreads();
    if (threadIdx.x == 0) {
        unsigned g = __hip_atomic_load(gen, __ATOMIC_ACQUIRE,
                                       __HIP_MEMORY_SCOPE_AGENT);
        unsigned arr = __hip_atomic_fetch_add(bar, 1u, __ATOMIC_ACQ_REL,
                                              __HIP_MEMORY_SCOPE_AGENT);
        if (arr == (unsigned)gridDim.x - 1u) {
            __hip_atomic_store(bar, 0u, __ATOMIC_RELAXED,
                               __HIP_MEMORY_SCOPE_AGENT);
            __hip_atomic_fetch_add(gen, 1u, __ATOMIC_RELEASE,
                                   __HIP_MEMORY_SCOPE_AGENT);
        } else {
            while (__hip_atomic_load(gen, __ATOMIC_ACQUIRE,
                                     __HIP_MEMORY_SCOPE_AGENT) == g) {
                __builtin_amdgcn_s_sleep(2);
            }
        }
    }
    __syncthreads();
    __builtin_amdgcn_fence(__ATOMIC_ACQUIRE, "agent");   // drop stale lines
}

union Frag { uint4 u[2]; v16h v; };

// 16x64 output strip (4 WMMA accumulators) of C = A(256xK) * W^T.
// A: global f16 row-major.  B: LDS panel (64 cols x 1024 K, stride LDSTRIDE).
// Both operands double-buffered across the 32 K-chunks.
__device__ __forceinline__ void gemm_strip(const h16* A, const h16* ldsW,
                                           int tile_m, int lane, v8f c[4]) {
    const int ml = lane & 15;   // A row / B column within tile
    const int h  = lane >> 4;   // lane half selects K sub-range
    const h16* arow  = A + (tile_m * 16 + ml) * 1024;
    const h16* bbase = ldsW + ml * LDSTRIDE + h * 8;

    Frag a_buf[2];
    Frag b_buf[2][4];

    // prologue: chunk 0
    a_buf[0].u[0] = *(const uint4*)(arow + h * 8);
    a_buf[0].u[1] = *(const uint4*)(arow + 16 + h * 8);
#pragma unroll
    for (int t = 0; t < 4; ++t) {
        const uint4* bp = (const uint4*)(bbase + t * 16 * LDSTRIDE);
        b_buf[0][t].u[0] = bp[0];
        b_buf[0][t].u[1] = bp[2];
    }

#pragma unroll 2
    for (int kc = 0; kc < 32; ++kc) {
        const int cur = kc & 1, nxt = cur ^ 1;
        if (kc < 31) {                       // prefetch chunk kc+1 (A + B)
            const int k1 = (kc + 1) * 32;
            a_buf[nxt].u[0] = *(const uint4*)(arow + k1 + h * 8);
            a_buf[nxt].u[1] = *(const uint4*)(arow + k1 + 16 + h * 8);
#pragma unroll
            for (int t = 0; t < 4; ++t) {
                const uint4* bp =
                    (const uint4*)(bbase + t * 16 * LDSTRIDE + k1);
                b_buf[nxt][t].u[0] = bp[0];
                b_buf[nxt][t].u[1] = bp[2];
            }
        }
#pragma unroll
        for (int t = 0; t < 4; ++t) {
            c[t] = __builtin_amdgcn_wmma_f32_16x16x32_f16(
                false, a_buf[cur].v, false, b_buf[cur][t].v,
                (short)0, c[t], false, false);
        }
    }
}

// --- setup kernels -----------------------------------------------------------

// Convert + transpose both recurrent weight matrices to f16, N-major.
__global__ __launch_bounds__(256) void conv_weights_kernel(
        const float* __restrict__ w1, const float* __restrict__ w2,
        h16* __restrict__ W1t, h16* __restrict__ W2t) {
    int i = blockIdx.x * blockDim.x + threadIdx.x;      // 1024*1024 threads
    if (i < 1024 * 1024) {
        int n = i & 1023, k = i >> 10;                  // coalesced reads over n
        W1t[n * 1024 + k] = (h16)w1[k * 1024 + n];
        W2t[n * 1024 + k] = (h16)w2[k * 1024 + n];
    }
}

// Initialize running state + zero the grid-barrier counters.
__global__ __launch_bounds__(256) void init_state_kernel(
        const float* __restrict__ ec3_last, const float* __restrict__ ec5_last,
        float* __restrict__ ec3s, float* __restrict__ ec5s,
        h16* __restrict__ ec3h, unsigned* __restrict__ counters) {
    int i = blockIdx.x * blockDim.x + threadIdx.x;      // 256*1024
    if (i < BS * ECNUM) {
        float e3 = ec3_last[i];
        ec3s[i] = e3;
        ec3h[i] = (h16)e3;
        ec5s[i] = ec5_last[i];
    }
    if (i == 0) { counters[0] = 0u; counters[1] = 0u; }
}

// drive[t][j] = sum_k exp(-(c_k - t)^2 / (2*sigma^2)) * wca3ca1[k][j]
__global__ __launch_bounds__(256) void drive_kernel(
        const float* __restrict__ wca3ca1, float* __restrict__ drive) {
    __shared__ float ca3[CA3NUM];
    const int t = blockIdx.x;                           // 0..99
    const float step = (float)t;
    for (int k = threadIdx.x; k < CA3NUM; k += 256) {
        float c  = (100.0f * (float)k) / 1023.0f;       // linspace(0,100,1024)
        float dx = c - step;
        ca3[k] = __expf(-dx * dx * (1.0f / 50.0f));     // sigma=5 -> /(25*2)
    }
    __syncthreads();
    for (int j = threadIdx.x; j < CA1NUM; j += 256) {
        float acc = 0.0f;
        for (int k = 0; k < CA3NUM; ++k)
            acc = fmaf(ca3[k], wca3ca1[k * CA1NUM + j], acc);
        drive[t * CA1NUM + j] = acc;
    }
}

// --- persistent RNN kernel ---------------------------------------------------
// NOTE: no __restrict__ here on purpose -- combined with the per-step compiler
// barrier it stops LICM from hoisting the (loop-invariant) weight loads out of
// the 100-step loop, which previously spilled ~4KB/lane to scratch.

__global__ __launch_bounds__(256) void rnn_persistent_kernel(
        const h16* W1t, const h16* W2t,
        const float* drive, const float* ca1bias, const int* cue,
        float* ec3s, float* ec5s, float* ca1s,
        h16* ec3h, h16* ca1h,
        float* ec3his, float* ec5his, float* ca1his,
        const float* wca1act, const float* actbias, float* actOut,
        unsigned* counters) {
    extern __shared__ h16 lds[];           // [2][64][LDSTRIDE]
    h16* ldsW1 = lds;
    h16* ldsW2 = lds + PANEL;

    unsigned* bar = counters;
    unsigned* gen = counters + 1;

    // Block b: tile_n = b>>1 (64-col N-panel), tile_m = (b&1)*8 + wave.
    const int lane    = threadIdx.x & 31;
    const int wave    = threadIdx.x >> 5;              // 0..7
    const int tile_n  = blockIdx.x >> 1;               // 0..15
    const int tile_m  = ((blockIdx.x & 1) << 3) + wave;// 0..15
    const int h  = lane >> 4;
    const int nl = lane & 15;

    // ---- stage both 64-col weight panels into LDS (once, reused 100x) ----
    {
        const h16* gW1 = W1t + tile_n * 64 * 1024;
        const h16* gW2 = W2t + tile_n * 64 * 1024;
        for (int idx = threadIdx.x; idx < 64 * 128; idx += 256) {
            const int col = idx >> 7;                  // 0..63
            const int seg = idx & 127;                 // k = seg*8
            const int go  = col * 1024 + seg * 8;
            const int lo  = col * LDSTRIDE + seg * 8;
            *(uint4*)&ldsW1[lo] = *(const uint4*)(gW1 + go);
            *(uint4*)&ldsW2[lo] = *(const uint4*)(gW2 + go);
        }
    }
    __syncthreads();

    for (int t = 0; t < TLEN; ++t) {
        asm volatile("" ::: "memory");     // no load motion across steps

        // ---- GEMM1 + epilogue: ca1 = relu(drive*(1+sigmoid(ec3@W1)) - bias)
        {
            v8f c[4] = {};
            gemm_strip(ec3h, ldsW1, tile_m, lane, c);
            const float* drive_t = drive + t * CA1NUM;
#pragma unroll
            for (int a = 0; a < 4; ++a) {
                const int col = tile_n * 64 + a * 16 + nl;
                const float d    = drive_t[col];
                const float bias = ca1bias[col];
#pragma unroll
                for (int i = 0; i < 8; ++i) {
                    const int row = tile_m * 16 + h * 8 + i;  // C/D: M = i + h*8
                    float r = d * (1.0f + sigmoidf_(c[a][i])) - bias;
                    r = fmaxf(r, 0.0f);
                    const int rc = row * CA1NUM + col;
                    ca1h[rc] = (h16)r;
                    if (t == TLEN - 1) ca1s[rc] = r;      // final ca1 only
                    if (row == 0) ca1his[t * CA1NUM + col] = r;
                }
            }
        }
        grid_sync(bar, gen);          // ca1h visible to all for GEMM2

        // ---- GEMM2 + epilogue: ec5 = f(ec5 + ca1@W2); ec3 = ec5*ec3 (+cue)
        {
            v8f c[4] = {};
            gemm_strip(ca1h, ldsW2, tile_m, lane, c);
            const int stim = (t == 16) ? 0 : ((t == 24) ? 1 : -1);
#pragma unroll
            for (int a = 0; a < 4; ++a) {
                const int col = tile_n * 64 + a * 16 + nl;
#pragma unroll
                for (int i = 0; i < 8; ++i) {
                    const int row = tile_m * 16 + h * 8 + i;
                    const int rc  = row * ECNUM + col;
                    float e5 = ec5s[rc] + c[a][i];        // 10.0 * TS == 1.0
                    e5 = 0.69f + 0.3f * sigmoidf_(4.0f * (e5 - 0.3f));
                    float e3 = e5 * ec3s[rc];
                    if (stim >= 0) {
                        if (cue[(row * 2 + stim) * 1024 + col])
                            e3 = 0.4f * e3 + 0.6f;
                    }
                    ec5s[rc] = e5;
                    ec3s[rc] = e3;
                    ec3h[rc] = (h16)e3;
                    if (row == 0) {
                        ec5his[t * ECNUM + col] = e5;
                        ec3his[t * ECNUM + col] = e3;
                    }
                }
            }
        }
        grid_sync(bar, gen);          // ec3h visible to all for next GEMM1
    }

    // ---- actCell = ca1 @ wca1act + actbias  (256x1024x2, threads 0..511)
    const int gtid = blockIdx.x * 256 + threadIdx.x;
    if (gtid < BS * 2) {
        const int b = gtid >> 1, a2 = gtid & 1;
        float acc = actbias[a2];
        for (int k = 0; k < CA1NUM; ++k)
            acc = fmaf(ca1s[b * CA1NUM + k], wca1act[k * 2 + a2], acc);
        actOut[gtid] = acc;
    }
}

// ---------------------------------------------------------------------------

extern "C" void kernel_launch(void* const* d_in, const int* in_sizes, int n_in,
                              void* d_out, int out_size, void* d_ws, size_t ws_size,
                              hipStream_t stream) {
    const int*   cue      = (const int*)d_in[0];
    const float* ec3_last = (const float*)d_in[1];
    const float* ec5_last = (const float*)d_in[2];
    // d_in[3] = ca1_last: initial carry never read by the scan body -> unused
    const float* ca1bias  = (const float*)d_in[4];
    const float* wca3ca1  = (const float*)d_in[5];
    const float* wec3ca1  = (const float*)d_in[6];
    const float* wca1ec5  = (const float*)d_in[7];
    const float* wca1act  = (const float*)d_in[8];
    const float* actbias  = (const float*)d_in[9];

    // Output layout (flat f32, reference return order):
    float* out    = (float*)d_out;
    float* actOut = out;                       // (256,2)    = 512
    float* ec3his = actOut + 512;              // (100,1024) = 102400
    float* ec5his = ec3his + TLEN * ECNUM;
    float* ca1his = ec5his + TLEN * ECNUM;
    float* ec3s   = ca1his + TLEN * CA1NUM;    // final ec3 (running state)
    float* ec5s   = ec3s + BS * ECNUM;         // final ec5 (running state)
    float* ca1s   = ec5s + BS * ECNUM;         // final ca1 (written at t=99)

    // Workspace layout (~5.4 MB)
    char*  ws    = (char*)d_ws;
    h16*   W1t   = (h16*)ws;                          // 1024*1024 f16 (N-major)
    h16*   W2t   = W1t + 1024 * 1024;                 // 1024*1024 f16
    float* drive = (float*)(W2t + 1024 * 1024);       // 100*1024 f32
    h16*   ec3h  = (h16*)(drive + TLEN * CA1NUM);     // 256*1024 f16
    h16*   ca1h  = ec3h + BS * ECNUM;                 // 256*1024 f16
    unsigned* counters = (unsigned*)(ca1h + BS * CA1NUM);   // bar, gen

    conv_weights_kernel<<<4096, 256, 0, stream>>>(wec3ca1, wca1ec5, W1t, W2t);
    init_state_kernel<<<1024, 256, 0, stream>>>(ec3_last, ec5_last,
                                                ec3s, ec5s, ec3h, counters);
    drive_kernel<<<TLEN, 256, 0, stream>>>(wca3ca1, drive);

    rnn_persistent_kernel<<<NBLK, 256, LDS_BYTES, stream>>>(
        W1t, W2t, drive, ca1bias, cue,
        ec3s, ec5s, ca1s, ec3h, ca1h,
        ec3his, ec5his, ca1his,
        wca1act, actbias, actOut, counters);
}
